// TransformerDecoder_75307956568273
// MI455X (gfx1250) — matrix operand order repaired and checked by reading, not compiled, not run on hardware
//
#include <hip/hip_runtime.h>
#include <math.h>

#define DEPTH 4
#define HEADS 16
#define DIM_  1024
#define DHEAD 64
#define MLP_  4096
#define BB    4
#define NN    1024
#define MM    1024
#define INNER (HEADS * DHEAD)   // 1024
#define ROWS  (BB * NN)         // 4096

typedef unsigned short bf16raw;
typedef __attribute__((ext_vector_type(16))) __bf16 v16bf;
typedef __attribute__((ext_vector_type(8)))  float  v8f;

union Frag16 { unsigned int u[8]; v16bf v; };

static __device__ __forceinline__ v8f zero8() {
  v8f z = {0.f, 0.f, 0.f, 0.f, 0.f, 0.f, 0.f, 0.f};
  return z;
}

// f32 -> bf16 round-to-nearest-even
static __device__ __forceinline__ bf16raw f2bf(float f) {
  union { float f; unsigned int u; } x; x.f = f;
  unsigned int r = x.u + 0x7FFFu + ((x.u >> 16) & 1u);
  return (bf16raw)(r >> 16);
}
static __device__ __forceinline__ unsigned int pack2(float lo, float hi) {
  return (unsigned int)f2bf(lo) | ((unsigned int)f2bf(hi) << 16);
}

// ---------------------------------------------------------------------------
// GEMM: C(Mx N) = A(bf16, MxK row-major) * W(f32, KxN row-major)
// Epilogues: GELU(+bias)->bf16 out, RESIDUAL: outf += acc + bias, else bf16 out
// Block tile 128x128, 8 waves of 32x64, K-step 32, bf16 WMMA w/ f32 accum.
// ---------------------------------------------------------------------------
template <bool GELU, bool RESIDUAL>
__global__ __launch_bounds__(256) void gemm_bf16w(
    const bf16raw* __restrict__ A, const float* __restrict__ Bw,
    const float* __restrict__ bias, bf16raw* __restrict__ outb,
    float* __restrict__ outf, int K, int Ncols)
{
  __shared__ bf16raw As[128][32];   // row-major activation tile
  __shared__ bf16raw Bs[128][32];   // TRANSPOSED weight tile: Bs[n][k]

  const int tid  = threadIdx.x;
  const int lane = tid & 31, wid = tid >> 5;
  const int col  = lane & 15, half = lane >> 4;
  const int m0 = blockIdx.y * 128, n0 = blockIdx.x * 128;
  const int wm = (wid & 3) * 32, wn = (wid >> 2) * 64;

  v8f acc[2][4];
  #pragma unroll
  for (int a = 0; a < 2; ++a)
    #pragma unroll
    for (int b = 0; b < 4; ++b) acc[a][b] = zero8();

  const int arow = tid >> 1, aseg = (tid & 1) * 16;   // A: 128 rows x 32 cols
  const int bk   = tid >> 3, bseg = (tid & 7) * 16;   // W: 32 rows x 128 cols

  for (int k0 = 0; k0 < K; k0 += 32) {
    // stage A tile (already bf16): 16 elements = 32B per thread
    const uint4* ga = (const uint4*)(A + (size_t)(m0 + arow) * K + k0 + aseg);
    *(uint4*)&As[arow][aseg]     = ga[0];
    *(uint4*)&As[arow][aseg + 8] = ga[1];
    // stage W tile: f32 -> bf16, transposed into Bs[n][k]
    const float* gb = Bw + (size_t)(k0 + bk) * Ncols + n0 + bseg;
    #pragma unroll
    for (int t = 0; t < 16; ++t) Bs[bseg + t][bk] = f2bf(gb[t]);
    if (k0 + 32 < K) {                 // hint next tiles toward the caches
      __builtin_prefetch(A + (size_t)(m0 + arow) * K + k0 + 32 + aseg, 0, 1);
      __builtin_prefetch(Bw + (size_t)(k0 + 32 + bk) * Ncols + n0 + bseg, 0, 1);
    }
    __syncthreads();

    // A fragment: lane = row, K pairs {0..7,16..23} (lanes 0-15) / {8..15,24..31}
    Frag16 af[2], bfr[4];
    #pragma unroll
    for (int tm = 0; tm < 2; ++tm)
      #pragma unroll
      for (int j = 0; j < 8; ++j) {
        const int kk = 2 * j + (j >= 4 ? 8 : 0) + 8 * half;
        af[tm].u[j] = *(const unsigned int*)&As[wm + tm * 16 + col][kk];
      }
    // B fragment: lane = col, K pairs {0..7,16..23} (lanes 0-15) / {8..15,24..31}
    #pragma unroll
    for (int tn = 0; tn < 4; ++tn)
      #pragma unroll
      for (int j = 0; j < 8; ++j) {
        const int kk = 2 * j + (j >= 4 ? 8 : 0) + 8 * half;
        bfr[tn].u[j] = *(const unsigned int*)&Bs[wn + tn * 16 + col][kk];
      }
    #pragma unroll
    for (int tm = 0; tm < 2; ++tm)
      #pragma unroll
      for (int tn = 0; tn < 4; ++tn)
        acc[tm][tn] = __builtin_amdgcn_wmma_f32_16x16x32_bf16(
            false, af[tm].v, false, bfr[tn].v, (short)0, acc[tm][tn], false, false);
    __syncthreads();
  }

  // Epilogue. C layout: VGPR r -> row r + 8*half, col = lane%16
  #pragma unroll
  for (int tn = 0; tn < 4; ++tn) {
    const int gc = n0 + wn + tn * 16 + col;
    const float bv = bias ? bias[gc] : 0.0f;
    #pragma unroll
    for (int tm = 0; tm < 2; ++tm) {
      #pragma unroll
      for (int r = 0; r < 8; ++r) {
        const int gr = m0 + wm + tm * 16 + r + 8 * half;
        float v = acc[tm][tn][r] + bv;
        if (GELU) v = 0.5f * v * (1.0f + erff(v * 0.70710678118654752f));
        if (RESIDUAL) outf[(size_t)gr * Ncols + gc] += v;
        else          outb[(size_t)gr * Ncols + gc] = f2bf(v);
      }
    }
  }
}

// ---------------------------------------------------------------------------
// Flash attention: one wave per (b, h, 16-query tile).
// S^T = K_chunk * Q^T so score C-layout == A-fragment layout for P (no lane
// transposes). Online softmax per lane; partner lanes (xor 16) share a query.
// ---------------------------------------------------------------------------
__global__ __launch_bounds__(256) void attn_kernel(
    const bf16raw* __restrict__ q, const bf16raw* __restrict__ k,
    const bf16raw* __restrict__ v, bf16raw* __restrict__ o)
{
  const int lane = threadIdx.x & 31;
  const int gw   = blockIdx.x * 8 + (threadIdx.x >> 5);
  const int QT   = NN / 16;
  const int qt = gw % QT;
  const int h  = (gw / QT) % HEADS;
  const int b  = gw / (QT * HEADS);
  const int col = lane & 15, half = lane >> 4;
  const float scale = 0.03125f;  // DIM^-0.5 = 1/32

  // Q^T (64 x 16) as two B-fragments: lane = query col, K = dim
  Frag16 bq[2];
  const bf16raw* qbase = q + (size_t)(b * NN + qt * 16 + col) * INNER + h * DHEAD;
  #pragma unroll
  for (int d = 0; d < 2; ++d)
    #pragma unroll
    for (int j = 0; j < 8; ++j)
      bq[d].u[j] = *(const unsigned int*)(qbase + d * 32 + 2 * j + (j >= 4 ? 8 : 0) + 8 * half);

  v8f acc[4];
  #pragma unroll
  for (int i = 0; i < 4; ++i) acc[i] = zero8();
  float m_run = -1e30f, l_run = 0.0f;

  for (int kc = 0; kc < MM; kc += 32) {
    // S^T tiles: ct[t] = K[kc+16t .. +15, :] @ Q^T  (accumulate over dim)
    v8f ct[2]; ct[0] = zero8(); ct[1] = zero8();
    #pragma unroll
    for (int t = 0; t < 2; ++t) {
      const bf16raw* kbase =
          k + (size_t)(b * MM + kc + t * 16 + col) * INNER + h * DHEAD;
      #pragma unroll
      for (int d = 0; d < 2; ++d) {
        Frag16 ak;
        #pragma unroll
        for (int j = 0; j < 8; ++j)
          ak.u[j] = *(const unsigned int*)(kbase + d * 32 + 2 * j +
                                           (j >= 4 ? 8 : 0) + 8 * half);
        ct[t] = __builtin_amdgcn_wmma_f32_16x16x32_bf16(
            false, ak.v, false, bq[d].v, (short)0, ct[t], false, false);
      }
    }
    // lane holds query `col`, keys {8*half + r} and {16 + 8*half + r}
    float s0[8], s1[8], cmax = -1e30f;
    #pragma unroll
    for (int r = 0; r < 8; ++r) {
      s0[r] = ct[0][r] * scale; s1[r] = ct[1][r] * scale;
      cmax = fmaxf(cmax, fmaxf(s0[r], s1[r]));
    }
    cmax = fmaxf(cmax, __shfl_xor(cmax, 16, 32));
    const float m_new = fmaxf(m_run, cmax);
    const float fac   = __expf(m_run - m_new);
    float p0[8], p1[8], lsum = 0.0f;
    #pragma unroll
    for (int r = 0; r < 8; ++r) {
      p0[r] = __expf(s0[r] - m_new);
      p1[r] = __expf(s1[r] - m_new);
      lsum += p0[r] + p1[r];
    }
    lsum += __shfl_xor(lsum, 16, 32);
    l_run = l_run * fac + lsum;
    m_run = m_new;

    // P (16 queries x 32 keys) directly in A-fragment layout
    Frag16 ap;
    #pragma unroll
    for (int j = 0; j < 4; ++j) ap.u[j] = pack2(p0[2 * j], p0[2 * j + 1]);
    #pragma unroll
    for (int j = 4; j < 8; ++j) ap.u[j] = pack2(p1[2 * (j - 4)], p1[2 * (j - 4) + 1]);

    // rescale accumulators: row r+8*half needs that query's factor
    #pragma unroll
    for (int r = 0; r < 8; ++r) {
      const float fr = __shfl(fac, r + 8 * half, 32);
      #pragma unroll
      for (int i = 0; i < 4; ++i) acc[i][r] *= fr;
    }
    // V chunk (32 keys x 64 dims) as four B fragments: lane = dim col, K = key
    const bf16raw* vbase =
        v + (size_t)(b * MM + kc + 8 * half) * INNER + h * DHEAD;
    #pragma unroll
    for (int i = 0; i < 4; ++i) {
      Frag16 bv;
      #pragma unroll
      for (int j = 0; j < 8; ++j) {
        const unsigned int lo = vbase[(size_t)(2 * j + (j >= 4 ? 8 : 0)) * INNER + i * 16 + col];
        const unsigned int hi = vbase[(size_t)(2 * j + (j >= 4 ? 8 : 0) + 1) * INNER + i * 16 + col];
        bv.u[j] = lo | (hi << 16);
      }
      acc[i] = __builtin_amdgcn_wmma_f32_16x16x32_bf16(
          false, ap.v, false, bv.v, (short)0, acc[i], false, false);
    }
  }

  bf16raw* obase = o + (size_t)(b * NN + qt * 16) * INNER + h * DHEAD;
  #pragma unroll
  for (int r = 0; r < 8; ++r) {
    const float inv = 1.0f / __shfl(l_run, r + 8 * half, 32);
    const int row = r + 8 * half;
    #pragma unroll
    for (int i = 0; i < 4; ++i)
      obase[(size_t)row * INNER + i * 16 + col] = f2bf(acc[i][r] * inv);
  }
}

// ---------------------------------------------------------------------------
// LayerNorm (one 256-thread block per 1024-elem row) -> bf16
// ---------------------------------------------------------------------------
__global__ __launch_bounds__(256) void ln_kernel(
    const float* __restrict__ x, const float* __restrict__ w,
    const float* __restrict__ bb, bf16raw* __restrict__ out)
{
  const int row = blockIdx.x, tid = threadIdx.x;
  __shared__ float rs[8], rs2[8];
  const float* xr = x + (size_t)row * DIM_;
  float v[4], s = 0.f, s2 = 0.f;
  #pragma unroll
  for (int i = 0; i < 4; ++i) {
    v[i] = xr[tid + i * 256];
    s += v[i]; s2 += v[i] * v[i];
  }
  #pragma unroll
  for (int off = 16; off > 0; off >>= 1) {
    s  += __shfl_xor(s,  off, 32);
    s2 += __shfl_xor(s2, off, 32);
  }
  if ((tid & 31) == 0) { rs[tid >> 5] = s; rs2[tid >> 5] = s2; }
  __syncthreads();
  float tot = 0.f, tot2 = 0.f;
  #pragma unroll
  for (int i = 0; i < 8; ++i) { tot += rs[i]; tot2 += rs2[i]; }
  const float mu  = tot * (1.0f / DIM_);
  const float var = tot2 * (1.0f / DIM_) - mu * mu;
  const float inv = rsqrtf(var + 1e-5f);
  #pragma unroll
  for (int i = 0; i < 4; ++i) {
    const int c = tid + i * 256;
    out[(size_t)row * DIM_ + c] = f2bf((v[i] - mu) * inv * w[c] + bb[c]);
  }
}

__global__ void copy_f32_kernel(const float* __restrict__ a,
                                float* __restrict__ o, int n) {
  int i = blockIdx.x * blockDim.x + threadIdx.x;
  if (i < n) o[i] = a[i];
}
__global__ void cvt_bf16_kernel(const float* __restrict__ a,
                                bf16raw* __restrict__ o, int n) {
  int i = blockIdx.x * blockDim.x + threadIdx.x;
  if (i < n) o[i] = f2bf(a[i]);
}

// ---------------------------------------------------------------------------
extern "C" void kernel_launch(void* const* d_in, const int* in_sizes, int n_in,
                              void* d_out, int out_size, void* d_ws, size_t ws_size,
                              hipStream_t stream)
{
  const float* x_in = (const float*)d_in[0];
  const float* m_in = (const float*)d_in[1];
  const float* Wq   = (const float*)d_in[2];
  const float* Wk   = (const float*)d_in[3];
  const float* Wv   = (const float*)d_in[4];
  const float* Wo   = (const float*)d_in[5];
  const float* bo   = (const float*)d_in[6];
  const float* ln1w = (const float*)d_in[7];
  const float* ln1b = (const float*)d_in[8];
  const float* W1   = (const float*)d_in[9];
  const float* b1   = (const float*)d_in[10];
  const float* W2   = (const float*)d_in[11];
  const float* b2   = (const float*)d_in[12];
  const float* ln2w = (const float*)d_in[13];
  const float* ln2b = (const float*)d_in[14];

  // workspace (bf16 activations): 6x8MB + 32MB = 80MB
  char* ws = (char*)d_ws;
  const size_t MB = (size_t)1 << 20;
  bf16raw* m_bf  = (bf16raw*)(ws + 0 * MB);
  bf16raw* h_bf  = (bf16raw*)(ws + 8 * MB);
  bf16raw* q_bf  = (bf16raw*)(ws + 16 * MB);
  bf16raw* k_bf  = (bf16raw*)(ws + 24 * MB);
  bf16raw* v_bf  = (bf16raw*)(ws + 32 * MB);
  bf16raw* ao_bf = (bf16raw*)(ws + 40 * MB);
  bf16raw* ff_bf = (bf16raw*)(ws + 48 * MB);

  float* x = (float*)d_out;                 // residual stream lives in d_out
  const int ne = BB * NN * DIM_;            // 4,194,304

  copy_f32_kernel<<<ne / 256, 256, 0, stream>>>(x_in, x, ne);
  cvt_bf16_kernel<<<ne / 256, 256, 0, stream>>>(m_in, m_bf, ne);

  const dim3 blk(256);
  for (int i = 0; i < DEPTH; ++i) {
    const size_t wqk = (size_t)i * DIM_ * INNER;
    // h = LN1(x)
    ln_kernel<<<ROWS, blk, 0, stream>>>(x, ln1w + i * DIM_, ln1b + i * DIM_, h_bf);
    // q/k/v projections
    gemm_bf16w<false, false><<<dim3(INNER / 128, ROWS / 128), blk, 0, stream>>>(
        h_bf, Wq + wqk, nullptr, q_bf, nullptr, DIM_, INNER);
    gemm_bf16w<false, false><<<dim3(INNER / 128, ROWS / 128), blk, 0, stream>>>(
        m_bf, Wk + wqk, nullptr, k_bf, nullptr, DIM_, INNER);
    gemm_bf16w<false, false><<<dim3(INNER / 128, ROWS / 128), blk, 0, stream>>>(
        m_bf, Wv + wqk, nullptr, v_bf, nullptr, DIM_, INNER);
    // attention
    attn_kernel<<<(BB * HEADS * (NN / 16)) / 8, blk, 0, stream>>>(
        q_bf, k_bf, v_bf, ao_bf);
    // x += attn_out @ Wo + bo
    gemm_bf16w<false, true><<<dim3(DIM_ / 128, ROWS / 128), blk, 0, stream>>>(
        ao_bf, Wo + (size_t)i * INNER * DIM_, bo + i * DIM_, nullptr, x, INNER, DIM_);
    // h2 = LN2(x)
    ln_kernel<<<ROWS, blk, 0, stream>>>(x, ln2w + i * DIM_, ln2b + i * DIM_, h_bf);
    // ff = gelu(h2 @ W1 + b1)
    gemm_bf16w<true, false><<<dim3(MLP_ / 128, ROWS / 128), blk, 0, stream>>>(
        h_bf, W1 + (size_t)i * DIM_ * MLP_, b1 + i * MLP_, ff_bf, nullptr, DIM_, MLP_);
    // x += ff @ W2 + b2
    gemm_bf16w<false, true><<<dim3(DIM_ / 128, ROWS / 128), blk, 0, stream>>>(
        ff_bf, W2 + (size_t)i * MLP_ * DIM_, b2 + i * DIM_, nullptr, x, MLP_, DIM_);
  }
}
